// Decoder_5446018532056
// MI455X (gfx1250) — matrix operand order repaired and checked by reading, not compiled, not run on hardware
//
#include <hip/hip_runtime.h>

// ---------------------------------------------------------------------------
// Types for CDNA5 WMMA (gfx1250, wave32)
// ---------------------------------------------------------------------------
typedef __attribute__((ext_vector_type(16))) __bf16 v16bf;
typedef __attribute__((ext_vector_type(8)))  __bf16 v8bf;
typedef __attribute__((ext_vector_type(8)))  float  v8f;

#define GLOBAL_AS __attribute__((address_space(1)))

// Model dimensions
#define B_    32
#define C_    128
#define M_    128
#define E_    512
#define H_    1024
#define PN_   256
#define AH_   128
#define KC_   31
#define P_    15
#define TTOK_ 512
#define TF_   256
#define IN0_  640                  // M + E
#define CUML_ (TTOK_ + 2 * P_)     // 542
#define KCAT1_ 1920                // (PN + E + M) + H = 896 + 1024
#define KCAT2_ 2688                // (H + E + M) + H = 1664 + 1024
#define G4H_  4096                 // 4*H
#define STLD_ 768                  // init_w2 output padded 641 -> 768 (x128)

// ---------------------------------------------------------------------------
// WMMA fragment loader (bf16 16x16x32, A and B both in contiguous k = 16*half + s order).
// o1/o2 are compile-time element offsets -> fold into load immediate offsets.
// Pointers are cast to the explicit global address space so the loop-carried
// PHIs still lower to global_load_b128 (FLAT would also burn DScnt/LDS path).
// ---------------------------------------------------------------------------
union V16u { v16bf v; v8bf h[2]; };

__device__ inline v16bf load_frag(const __bf16* __restrict__ p, int o1, int o2) {
  GLOBAL_AS const __bf16* gp = (GLOBAL_AS const __bf16*)p;
  V16u u;
  u.h[0] = *(GLOBAL_AS const v8bf*)(gp + o1);
  u.h[1] = *(GLOBAL_AS const v8bf*)(gp + o2);
  return u.v;
}

// ---------------------------------------------------------------------------
// Generic linear: out = act(X(MxK) * W(NxK)^T + bias), fp32 and/or bf16 out.
// 4 waves/block; each wave computes a 32x32 register tile (MT=2 x NT=2).
// k-loop: ping-pong double buffer, unrolled by 2 (K % 64 == 0 at all call
// sites), with two independent accumulator chains so dependent WMMAs are 8
// apart (clears the bf16 WMMA RAW hazard window without v_nops), and loads
// write directly into the consumed registers (no cross-buffer moves).
// Requires: M % 32 == 0, N covered exactly by 128-wide block slabs, K % 64 == 0.
// ---------------------------------------------------------------------------
#define LIN_WAVES 4
#define LIN_MT    2
#define LIN_NT    2

#define WMMA_BF16(acc, a, b) \
  acc = __builtin_amdgcn_wmma_f32_16x16x32_bf16(false, a, false, b, (short)0, acc, false, false)

__global__ __launch_bounds__(LIN_WAVES * 32) void wmma_linear_kernel(
    const __bf16* __restrict__ X, const __bf16* __restrict__ W,
    const float* __restrict__ bias, float* __restrict__ outF,
    __bf16* __restrict__ outB, int M, int N, int K,
    int ldx, int ldw, int ldo, int act) {
  int lane = threadIdx.x & 31;
  int wave = threadIdx.x >> 5;
  int half = lane >> 4;
  int l15  = lane & 15;
  int m0   = blockIdx.y * (LIN_MT * 16);
  int n0   = (blockIdx.x * LIN_WAVES + wave) * (LIN_NT * 16);

  // Per-lane tile-row pointers; advanced by 64 elems per unrolled iteration.
  const __bf16* pa[LIN_MT];
  const __bf16* pb[LIN_NT];
#pragma unroll
  for (int i = 0; i < LIN_MT; ++i)
    pa[i] = X + (size_t)(m0 + i * 16 + l15) * ldx + half * 16;
#pragma unroll
  for (int j = 0; j < LIN_NT; ++j)
    pb[j] = W + (size_t)(n0 + j * 16 + l15) * ldw + half * 16;

  v8f acc0[LIN_MT][LIN_NT] = {};
  v8f acc1[LIN_MT][LIN_NT] = {};

  v16bf a0[LIN_MT], b0[LIN_NT], a1[LIN_MT], b1[LIN_NT];
#pragma unroll
  for (int i = 0; i < LIN_MT; ++i) {
    a0[i] = load_frag(pa[i], 0, 8);
    a1[i] = load_frag(pa[i], 32, 40);
  }
#pragma unroll
  for (int j = 0; j < LIN_NT; ++j) {
    b0[j] = load_frag(pb[j], 0, 8);
    b1[j] = load_frag(pb[j], 32, 40);
  }

  int kIter = K >> 5;            // even at all call sites
  for (int it = 0; it + 2 < kIter; it += 2) {
    // consume even buffer, then refill it for iteration it+2
#pragma unroll
    for (int j = 0; j < LIN_NT; ++j)
#pragma unroll
      for (int i = 0; i < LIN_MT; ++i) WMMA_BF16(acc0[i][j], a0[i], b0[j]);
#pragma unroll
    for (int i = 0; i < LIN_MT; ++i) pa[i] += 64;
#pragma unroll
    for (int j = 0; j < LIN_NT; ++j) pb[j] += 64;
#pragma unroll
    for (int i = 0; i < LIN_MT; ++i) a0[i] = load_frag(pa[i], 0, 8);
#pragma unroll
    for (int j = 0; j < LIN_NT; ++j) b0[j] = load_frag(pb[j], 0, 8);
    // consume odd buffer, then refill it for iteration it+3
#pragma unroll
    for (int j = 0; j < LIN_NT; ++j)
#pragma unroll
      for (int i = 0; i < LIN_MT; ++i) WMMA_BF16(acc1[i][j], a1[i], b1[j]);
#pragma unroll
    for (int i = 0; i < LIN_MT; ++i) a1[i] = load_frag(pa[i], 32, 40);
#pragma unroll
    for (int j = 0; j < LIN_NT; ++j) b1[j] = load_frag(pb[j], 32, 40);
  }
  // tail: last two k-steps, no refills
#pragma unroll
  for (int j = 0; j < LIN_NT; ++j)
#pragma unroll
    for (int i = 0; i < LIN_MT; ++i) WMMA_BF16(acc0[i][j], a0[i], b0[j]);
#pragma unroll
  for (int j = 0; j < LIN_NT; ++j)
#pragma unroll
    for (int i = 0; i < LIN_MT; ++i) WMMA_BF16(acc1[i][j], a1[i], b1[j]);

  // C/D layout: VGPR r -> lanes 0-15: M=r, lanes 16-31: M=r+8; N = lane&15.
#pragma unroll
  for (int j = 0; j < LIN_NT; ++j) {
    int n = n0 + j * 16 + l15;
    float bv = bias ? bias[n] : 0.0f;
#pragma unroll
    for (int i = 0; i < LIN_MT; ++i) {
#pragma unroll
      for (int r = 0; r < 8; ++r) {
        int row = m0 + i * 16 + half * 8 + r;
        float v = acc0[i][j][r] + acc1[i][j][r] + bv;
        if (act) v = fmaxf(v, 0.0f);
        if (outF) outF[(size_t)row * ldo + n] = v;
        if (outB) outB[(size_t)row * ldo + n] = (__bf16)v;
      }
    }
  }
}

// ---------------------------------------------------------------------------
// Conversion / packing kernels
// ---------------------------------------------------------------------------
__global__ void f32_to_bf16_kernel(const float* __restrict__ s,
                                   __bf16* __restrict__ d, long n) {
  long i = (long)blockIdx.x * blockDim.x + threadIdx.x;
  if (i < n) d[i] = (__bf16)s[i];
}

// Convert (srcRows x cols) f32 -> (dstRows x cols) bf16, zero rows >= srcRows.
__global__ void f32_to_bf16_rowpad_kernel(const float* __restrict__ s,
                                          __bf16* __restrict__ d,
                                          int srcRows, int dstRows, int cols) {
  long i = (long)blockIdx.x * blockDim.x + threadIdx.x;
  if (i >= (long)dstRows * cols) return;
  int r = (int)(i / cols);
  d[i] = (__bf16)((r < srcRows) ? s[i] : 0.0f);
}

__global__ void pad_f32_kernel(const float* __restrict__ s,
                               float* __restrict__ d, int n, int ntot) {
  int i = blockIdx.x * blockDim.x + threadIdx.x;
  if (i < ntot) d[i] = (i < n) ? s[i] : 0.0f;
}

__global__ void transpose_bf16_kernel(const float* __restrict__ s,
                                      __bf16* __restrict__ d, int R, int Cc) {
  // s is (R x Cc); d is (Cc x R): d[c][r] = s[r][c]
  long i = (long)blockIdx.x * blockDim.x + threadIdx.x;
  if (i >= (long)R * Cc) return;
  int r = (int)(i / Cc), c = (int)(i % Cc);
  d[(size_t)c * R + r] = (__bf16)s[i];
}

__global__ void cat2_bf16_kernel(const float* __restrict__ A, int K1,
                                 const float* __restrict__ Bm, int K2,
                                 __bf16* __restrict__ d, long total) {
  long i = (long)blockIdx.x * blockDim.x + threadIdx.x;
  if (i >= total) return;
  int K = K1 + K2;
  long n = i / K; int k = (int)(i % K);
  d[i] = (__bf16)((k < K1) ? A[n * K1 + k] : Bm[n * K2 + (k - K1)]);
}

__global__ void add_vec_kernel(const float* __restrict__ a,
                               const float* __restrict__ b,
                               float* __restrict__ o, int n) {
  int i = blockIdx.x * blockDim.x + threadIdx.x;
  if (i < n) o[i] = a[i] + b[i];
}

__global__ void zero_f32_kernel(float* __restrict__ p, long n) {
  long i = (long)blockIdx.x * blockDim.x + threadIdx.x;
  if (i < n) p[i] = 0.0f;
}

__global__ void pack_first_kernel(const float* __restrict__ meta,
                                  const float* __restrict__ tokens,
                                  __bf16* __restrict__ F) {
  int i = blockIdx.x * blockDim.x + threadIdx.x;
  if (i >= B_ * IN0_) return;
  int b = i / IN0_, c = i % IN0_;
  float v = (c < M_) ? meta[b * M_ + c] : tokens[(size_t)b * E_ + (c - M_)];
  F[i] = (__bf16)v;
}

__global__ void init_post_kernel(const float* __restrict__ st,
                                 float* __restrict__ cum,
                                 float* __restrict__ ctx) {
  int i = blockIdx.x * blockDim.x + threadIdx.x;
  if (i < B_ * CUML_) {
    int b = i / CUML_, j = i % CUML_;
    cum[i] = (j < P_) ? fabsf(st[b * STLD_ + C_]) : 0.0f;
  } else {
    int k = i - B_ * CUML_;
    if (k < B_ * E_) {
      int b = k / E_, e = k % E_;
      ctx[k] = st[b * STLD_ + C_ + 1 + e];
    }
  }
}

__global__ void pack_x0_kernel(const float* __restrict__ st,
                               const float* __restrict__ tgt,
                               const float* __restrict__ meta,
                               __bf16* __restrict__ X0) {
  int i = blockIdx.x * blockDim.x + threadIdx.x;
  if (i >= TF_ * B_ * 256) return;
  int r = i >> 8, c = i & 255;
  int t = r >> 5, b = r & 31;
  float v;
  if (c < C_)
    v = (t == 0) ? st[b * STLD_ + c] : tgt[((size_t)(t - 1) * B_ + b) * C_ + c];
  else
    v = meta[b * M_ + (c - C_)];
  X0[i] = (__bf16)v;
}

// Xs1 (32 x 1920): [0,256) pre_t | [256,768) ctx | [768,896) meta | [896,1920) h1
__global__ void pack_step1_kernel(const __bf16* __restrict__ pre_t,
                                  const float* __restrict__ ctx,
                                  const __bf16* __restrict__ metab,
                                  const float* __restrict__ h1,
                                  __bf16* __restrict__ Xs) {
  int i = blockIdx.x * blockDim.x + threadIdx.x;
  if (i >= B_ * KCAT1_) return;
  int b = i / KCAT1_, c = i % KCAT1_;
  __bf16 v;
  if (c < 256)       v = pre_t[b * 256 + c];
  else if (c < 768)  v = (__bf16)ctx[b * E_ + (c - 256)];
  else if (c < 896)  v = metab[b * M_ + (c - 768)];
  else               v = (__bf16)h1[b * H_ + (c - 896)];
  Xs[i] = v;
}

// Xs2 (32 x 2688): [0,1024) hs_t | [1024,1536) ctxs_t | [1536,1664) meta | [1664,2688) h2
__global__ void pack_step2_kernel(const float* __restrict__ hs_t,
                                  const float* __restrict__ ctxs_t,
                                  const __bf16* __restrict__ metab,
                                  const float* __restrict__ h2,
                                  __bf16* __restrict__ Xs) {
  int i = blockIdx.x * blockDim.x + threadIdx.x;
  if (i >= B_ * KCAT2_) return;
  int b = i / KCAT2_, c = i % KCAT2_;
  __bf16 v;
  if (c < 1024)      v = (__bf16)hs_t[b * H_ + c];
  else if (c < 1536) v = (__bf16)ctxs_t[b * E_ + (c - 1024)];
  else if (c < 1664) v = metab[b * M_ + (c - 1536)];
  else               v = (__bf16)h2[b * H_ + (c - 1664)];
  Xs[i] = v;
}

// Fill X3 (8192 x 1664) columns [1024,1664) = [ctxs | meta] for all rows.
__global__ void x3_fill_kernel(const float* __restrict__ ctxs,
                               const __bf16* __restrict__ metab,
                               __bf16* __restrict__ X3) {
  long i = (long)blockIdx.x * blockDim.x + threadIdx.x;
  if (i >= (long)TF_ * B_ * (E_ + M_)) return;
  int r = (int)(i / (E_ + M_)), c = (int)(i % (E_ + M_));
  int b = r & 31;
  __bf16 v = (c < E_) ? (__bf16)ctxs[(size_t)r * E_ + c] : metab[b * M_ + (c - E_)];
  X3[(size_t)r * (H_ + E_ + M_) + H_ + c] = v;
}

// ---------------------------------------------------------------------------
// LSTM cell elementwise: z(B x 4H) [i|f|g|o], updates c,h in place.
// ---------------------------------------------------------------------------
__global__ void lstm_cell_kernel(const float* __restrict__ z,
                                 float* __restrict__ c, float* __restrict__ h,
                                 __bf16* __restrict__ hbf, int hbfStride,
                                 float* __restrict__ hsave) {
  int i = blockIdx.x * blockDim.x + threadIdx.x;
  if (i >= B_ * H_) return;
  int b = i >> 10, j = i & (H_ - 1);
  const float* zr = z + (size_t)b * G4H_;
  float iv = zr[j], fv = zr[H_ + j], gv = zr[2 * H_ + j], ov = zr[3 * H_ + j];
  float ig = 1.0f / (1.0f + __expf(-iv));
  float fg = 1.0f / (1.0f + __expf(-fv));
  float og = 1.0f / (1.0f + __expf(-ov));
  float cn = fg * c[i] + ig * tanhf(gv);
  float hn = og * tanhf(cn);
  c[i] = cn;
  h[i] = hn;
  if (hbf)   hbf[(size_t)b * hbfStride + j] = (__bf16)hn;
  if (hsave) hsave[i] = hn;
}

// ---------------------------------------------------------------------------
// Attention score: one wave per (b, t) pair; lanes cover AH=128 (4 each).
// ---------------------------------------------------------------------------
__global__ __launch_bounds__(256) void score_kernel(
    const float* __restrict__ cum, const float* __restrict__ conv,
    const float* __restrict__ mem, const float* __restrict__ q,
    const float* __restrict__ v, const unsigned char* __restrict__ mask,
    float* __restrict__ score) {
  int gw   = (blockIdx.x * blockDim.x + threadIdx.x) >> 5;
  int lane = threadIdx.x & 31;
  int t = gw & (TTOK_ - 1);
  int b = gw >> 9;
  const float* cm = cum + b * CUML_ + t;
  float acc = 0.0f;
#pragma unroll
  for (int hh = 0; hh < 4; ++hh) {
    int h = lane + hh * 32;
    const float* cw = conv + h * KC_;
    float loc = 0.0f;
    for (int j = 0; j < KC_; ++j) loc += cm[j] * cw[j];
    float s = tanhf(loc + mem[((size_t)t * B_ + b) * AH_ + h] + q[b * AH_ + h]);
    acc += v[h] * s;
  }
#pragma unroll
  for (int o = 16; o > 0; o >>= 1) acc += __shfl_xor(acc, o, 32);
  if (lane == 0) {
    bool mk = mask[b * TTOK_ + t] != 0;
    score[b * TTOK_ + t] = mk ? acc : -3.4028234663852886e38f;
  }
}

// Softmax over tokens (per batch row) + cumulative-attention update (in place).
__global__ __launch_bounds__(256) void softmax_cum_kernel(
    float* __restrict__ sc, float* __restrict__ cum) {
  __shared__ float red[256];
  int b = blockIdx.x, tid = threadIdx.x;
  float m = -3.4e38f;
  for (int t = tid; t < TTOK_; t += 256) m = fmaxf(m, sc[b * TTOK_ + t]);
  red[tid] = m; __syncthreads();
  for (int s = 128; s > 0; s >>= 1) {
    if (tid < s) red[tid] = fmaxf(red[tid], red[tid + s]);
    __syncthreads();
  }
  float mx = red[0]; __syncthreads();
  float sum = 0.0f;
  for (int t = tid; t < TTOK_; t += 256) {
    float e = __expf(sc[b * TTOK_ + t] - mx);
    sc[b * TTOK_ + t] = e;
    sum += e;
  }
  red[tid] = sum; __syncthreads();
  for (int s = 128; s > 0; s >>= 1) {
    if (tid < s) red[tid] += red[tid + s];
    __syncthreads();
  }
  float inv = 1.0f / red[0];
  for (int t = tid; t < TTOK_; t += 256) {
    float a = sc[b * TTOK_ + t] * inv;
    sc[b * TTOK_ + t] = a;
    cum[b * CUML_ + P_ + t] += a;
  }
}

// ctx[b,e] = sum_t align[b,t] * tokens[t,b,e]
__global__ __launch_bounds__(256) void ctx_kernel(
    const float* __restrict__ align, const float* __restrict__ tokens,
    float* __restrict__ ctx, float* __restrict__ ctxs_t) {
  int b = blockIdx.x, tid = threadIdx.x;
  for (int e = tid; e < E_; e += 256) {
    float s = 0.0f;
    for (int t = 0; t < TTOK_; ++t)
      s += align[b * TTOK_ + t] * tokens[((size_t)t * B_ + b) * E_ + e];
    ctx[b * E_ + e] = s;
    ctxs_t[b * E_ + e] = s;
  }
}

// ---------------------------------------------------------------------------
// Host orchestration
// ---------------------------------------------------------------------------
#define GRID1D(n) dim3((unsigned)(((long)(n) + 255) / 256)), dim3(256)

extern "C" void kernel_launch(void* const* d_in, const int* in_sizes, int n_in,
                              void* d_out, int out_size, void* d_ws,
                              size_t ws_size, hipStream_t stream) {
  (void)in_sizes; (void)n_in; (void)out_size; (void)ws_size;

  const float* tokens    = (const float*)d_in[0];
  const float* meta      = (const float*)d_in[1];
  const float* tgt       = (const float*)d_in[2];
  const unsigned char* mask = (const unsigned char*)d_in[3];
  const float* init_w1   = (const float*)d_in[4];
  const float* init_b1   = (const float*)d_in[5];
  const float* init_w2   = (const float*)d_in[6];
  const float* init_b2   = (const float*)d_in[7];
  const float* pre_w1    = (const float*)d_in[8];
  const float* pre_b1    = (const float*)d_in[9];
  const float* pre_w2    = (const float*)d_in[10];
  const float* pre_b2    = (const float*)d_in[11];
  const float* l1_wih    = (const float*)d_in[12];
  const float* l1_whh    = (const float*)d_in[13];
  const float* l1_bih    = (const float*)d_in[14];
  const float* l1_bhh    = (const float*)d_in[15];
  const float* l2_wih    = (const float*)d_in[16];
  const float* l2_whh    = (const float*)d_in[17];
  const float* l2_bih    = (const float*)d_in[18];
  const float* l2_bhh    = (const float*)d_in[19];
  const float* attn_wq   = (const float*)d_in[20];
  const float* attn_bq   = (const float*)d_in[21];
  const float* attn_wm   = (const float*)d_in[22];
  const float* attn_conv = (const float*)d_in[23];
  const float* attn_v    = (const float*)d_in[24];
  const float* out_w     = (const float*)d_in[25];
  const float* out_b     = (const float*)d_in[26];
  // d_in[27] stop_w, d_in[28] stop_b: unused by the returned output.

  // --- workspace carve (256B aligned) ---
  char* wp = (char*)d_ws;
  auto carve = [&](size_t elems, size_t esz) -> char* {
    char* r = wp;
    wp += ((elems * esz + 255) & ~(size_t)255);
    return r;
  };
  __bf16* wcat1  = (__bf16*)carve((size_t)G4H_ * KCAT1_, 2);
  __bf16* wcat2  = (__bf16*)carve((size_t)G4H_ * KCAT2_, 2);
  float*  bias1  = (float*)carve(G4H_, 4);
  float*  bias2  = (float*)carve(G4H_, 4);
  __bf16* wpre1  = (__bf16*)carve((size_t)PN_ * (C_ + M_), 2);
  __bf16* wpre2  = (__bf16*)carve((size_t)PN_ * PN_, 2);
  __bf16* wq     = (__bf16*)carve((size_t)AH_ * H_, 2);
  __bf16* wmT    = (__bf16*)carve((size_t)AH_ * E_, 2);
  __bf16* wi1    = (__bf16*)carve((size_t)IN0_ * IN0_, 2);
  __bf16* wi2    = (__bf16*)carve((size_t)STLD_ * IN0_, 2);   // padded rows
  float*  bi2p   = (float*)carve(STLD_, 4);                   // padded bias
  __bf16* wout   = (__bf16*)carve((size_t)C_ * (H_ + E_ + M_), 2);
  __bf16* tokbf  = (__bf16*)carve((size_t)TTOK_ * B_ * E_, 2);
  __bf16* metab  = (__bf16*)carve((size_t)B_ * M_, 2);
  __bf16* firstb = (__bf16*)carve((size_t)B_ * IN0_, 2);
  __bf16* hmidb  = (__bf16*)carve((size_t)B_ * IN0_, 2);
  float*  st     = (float*)carve((size_t)B_ * STLD_, 4);
  __bf16* X0     = (__bf16*)carve((size_t)TF_ * B_ * 256, 2);
  __bf16* X1     = (__bf16*)carve((size_t)TF_ * B_ * 256, 2);
  __bf16* prebf  = (__bf16*)carve((size_t)TF_ * B_ * PN_, 2);
  float*  memf   = (float*)carve((size_t)TTOK_ * B_ * AH_, 4);
  float*  cum    = (float*)carve((size_t)B_ * CUML_, 4);
  float*  ctx    = (float*)carve((size_t)B_ * E_, 4);
  float*  h1     = (float*)carve((size_t)B_ * H_, 4);
  float*  c1     = (float*)carve((size_t)B_ * H_, 4);
  float*  h2     = (float*)carve((size_t)B_ * H_, 4);
  float*  c2     = (float*)carve((size_t)B_ * H_, 4);
  __bf16* h1bf   = (__bf16*)carve((size_t)B_ * H_, 2);
  float*  q      = (float*)carve((size_t)B_ * AH_, 4);
  float*  score  = (float*)carve((size_t)B_ * TTOK_, 4);
  float*  z      = (float*)carve((size_t)B_ * G4H_, 4);
  __bf16* Xs1    = (__bf16*)carve((size_t)B_ * KCAT1_, 2);
  __bf16* Xs2    = (__bf16*)carve((size_t)B_ * KCAT2_, 2);
  float*  hs     = (float*)carve((size_t)TF_ * B_ * H_, 4);
  float*  ctxs   = (float*)carve((size_t)TF_ * B_ * E_, 4);
  __bf16* X3     = (__bf16*)carve((size_t)TF_ * B_ * (H_ + E_ + M_), 2);

  auto linear = [&](const __bf16* X, const __bf16* W, const float* bias,
                    float* oF, __bf16* oB, int Mm, int Nn, int Kk,
                    int ldx, int ldw, int ldo, int act) {
    dim3 g((Nn + LIN_WAVES * LIN_NT * 16 - 1) / (LIN_WAVES * LIN_NT * 16),
           (Mm + LIN_MT * 16 - 1) / (LIN_MT * 16));
    wmma_linear_kernel<<<g, LIN_WAVES * 32, 0, stream>>>(
        X, W, bias, oF, oB, Mm, Nn, Kk, ldx, ldw, ldo, act);
  };

  // --- one-time weight conversion to bf16 ---
  cat2_bf16_kernel<<<GRID1D((long)G4H_ * KCAT1_), 0, stream>>>(
      l1_wih, 896, l1_whh, H_, wcat1, (long)G4H_ * KCAT1_);
  cat2_bf16_kernel<<<GRID1D((long)G4H_ * KCAT2_), 0, stream>>>(
      l2_wih, 1664, l2_whh, H_, wcat2, (long)G4H_ * KCAT2_);
  add_vec_kernel<<<GRID1D(G4H_), 0, stream>>>(l1_bih, l1_bhh, bias1, G4H_);
  add_vec_kernel<<<GRID1D(G4H_), 0, stream>>>(l2_bih, l2_bhh, bias2, G4H_);
  f32_to_bf16_kernel<<<GRID1D((long)PN_ * 256), 0, stream>>>(pre_w1, wpre1, (long)PN_ * 256);
  f32_to_bf16_kernel<<<GRID1D((long)PN_ * PN_), 0, stream>>>(pre_w2, wpre2, (long)PN_ * PN_);
  f32_to_bf16_kernel<<<GRID1D((long)AH_ * H_), 0, stream>>>(attn_wq, wq, (long)AH_ * H_);
  transpose_bf16_kernel<<<GRID1D((long)E_ * AH_), 0, stream>>>(attn_wm, wmT, E_, AH_);
  f32_to_bf16_kernel<<<GRID1D((long)IN0_ * IN0_), 0, stream>>>(init_w1, wi1, (long)IN0_ * IN0_);
  f32_to_bf16_rowpad_kernel<<<GRID1D((long)STLD_ * IN0_), 0, stream>>>(
      init_w2, wi2, 641, STLD_, IN0_);
  pad_f32_kernel<<<GRID1D(STLD_), 0, stream>>>(init_b2, bi2p, 641, STLD_);
  f32_to_bf16_kernel<<<GRID1D((long)C_ * (H_ + E_ + M_)), 0, stream>>>(
      out_w, wout, (long)C_ * (H_ + E_ + M_));
  f32_to_bf16_kernel<<<GRID1D((long)TTOK_ * B_ * E_), 0, stream>>>(
      tokens, tokbf, (long)TTOK_ * B_ * E_);
  f32_to_bf16_kernel<<<GRID1D((long)B_ * M_), 0, stream>>>(meta, metab, (long)B_ * M_);

  // --- state init ---
  zero_f32_kernel<<<GRID1D((long)B_ * H_), 0, stream>>>(h1, (long)B_ * H_);
  zero_f32_kernel<<<GRID1D((long)B_ * H_), 0, stream>>>(c1, (long)B_ * H_);
  zero_f32_kernel<<<GRID1D((long)B_ * H_), 0, stream>>>(h2, (long)B_ * H_);
  zero_f32_kernel<<<GRID1D((long)B_ * H_), 0, stream>>>(c2, (long)B_ * H_);

  // --- init network: st = (relu(first @ w1^T + b1)) @ w2^T + b2 (N padded) ---
  pack_first_kernel<<<GRID1D(B_ * IN0_), 0, stream>>>(meta, tokens, firstb);
  linear(firstb, wi1, init_b1, nullptr, hmidb, B_, IN0_, IN0_, IN0_, IN0_, IN0_, 1);
  linear(hmidb, wi2, bi2p, st, nullptr, B_, STLD_, IN0_, IN0_, IN0_, STLD_, 0);
  init_post_kernel<<<GRID1D(B_ * CUML_ + B_ * E_), 0, stream>>>(st, cum, ctx);

  // --- pre-net over all frames ---
  pack_x0_kernel<<<GRID1D((long)TF_ * B_ * 256), 0, stream>>>(st, tgt, meta, X0);
  linear(X0, wpre1, pre_b1, nullptr, X1, TF_ * B_, PN_, 256, 256, 256, PN_, 1);
  linear(X1, wpre2, pre_b2, nullptr, prebf, TF_ * B_, PN_, PN_, PN_, PN_, PN_, 1);

  // --- mem = tokens @ attn_wm ---
  linear(tokbf, wmT, nullptr, memf, nullptr, TTOK_ * B_, AH_, E_, E_, E_, AH_, 0);

  // --- phase 1: LSTM1 + attention (sequential) ---
  for (int t = 0; t < TF_; ++t) {
    pack_step1_kernel<<<GRID1D(B_ * KCAT1_), 0, stream>>>(
        prebf + (size_t)t * B_ * PN_, ctx, metab, h1, Xs1);
    linear(Xs1, wcat1, bias1, z, nullptr, B_, G4H_, KCAT1_, KCAT1_, KCAT1_, G4H_, 0);
    lstm_cell_kernel<<<GRID1D(B_ * H_), 0, stream>>>(
        z, c1, h1, h1bf, H_, hs + (size_t)t * B_ * H_);
    linear(h1bf, wq, attn_bq, q, nullptr, B_, AH_, H_, H_, H_, AH_, 0);
    score_kernel<<<dim3(2048), dim3(256), 0, stream>>>(
        cum, attn_conv, memf, q, attn_v, mask, score);
    softmax_cum_kernel<<<dim3(B_), dim3(256), 0, stream>>>(score, cum);
    ctx_kernel<<<dim3(B_), dim3(256), 0, stream>>>(
        score, tokens, ctx, ctxs + (size_t)t * B_ * E_);
  }

  // --- phase 2: LSTM2 (sequential), h2 written directly into X3 ---
  x3_fill_kernel<<<GRID1D((long)TF_ * B_ * (E_ + M_)), 0, stream>>>(ctxs, metab, X3);
  for (int t = 0; t < TF_; ++t) {
    pack_step2_kernel<<<GRID1D(B_ * KCAT2_), 0, stream>>>(
        hs + (size_t)t * B_ * H_, ctxs + (size_t)t * B_ * E_, metab, h2, Xs2);
    linear(Xs2, wcat2, bias2, z, nullptr, B_, G4H_, KCAT2_, KCAT2_, KCAT2_, G4H_, 0);
    lstm_cell_kernel<<<GRID1D(B_ * H_), 0, stream>>>(
        z, c2, h2, X3 + (size_t)t * B_ * (H_ + E_ + M_), H_ + E_ + M_, nullptr);
  }

  // --- final projection: out = X3 @ out_w^T + out_b ---
  linear(X3, wout, out_b, (float*)d_out, nullptr, TF_ * B_, C_,
         H_ + E_ + M_, H_ + E_ + M_, H_ + E_ + M_, C_, 0);
}
